// PhysMamba_64802466562249
// MI455X (gfx1250) — matrix operand order repaired and checked by reading, not compiled, not run on hardware
//
#include <hip/hip_runtime.h>
#include <hip/hip_bf16.h>

typedef __attribute__((ext_vector_type(16))) _Float16 v16h;
typedef __attribute__((ext_vector_type(8)))  float    v8f;

#define BB_ 32
#define NN_ 4096
#define CC_ 96
#define HH_ 192
static constexpr float EPSV = 1e-5f;

__device__ inline v16h pack16(const float* f) {
  v16h a;
#pragma unroll
  for (int j = 0; j < 16; ++j) a[j] = (_Float16)f[j];
  return a;
}

// ---------------------------------------------------------------------------
// Kernel 1: fc1 GEMM ([B*N,C] x [H,C]^T) + BN1 + ReLU, output TRANSPOSED to
// h_t[B, H, N] so the temporal FFT reads contiguous sequences.
// One wave32 computes one 16x16 output tile via 3x v_wmma_f32_16x16x32_f16.
// ---------------------------------------------------------------------------
__global__ __launch_bounds__(256)
void gemm1_bn_relu(const float* __restrict__ x, const float* __restrict__ w1,
                   const float* __restrict__ g, const float* __restrict__ be,
                   const float* __restrict__ mu, const float* __restrict__ var,
                   float* __restrict__ h_t) {
  const int lane    = threadIdx.x & 31;
  const int wg      = blockIdx.x * 8 + (threadIdx.x >> 5);
  const int rowTile = wg / (HH_ / 16);
  const int colTile = wg % (HH_ / 16);
  const int row0    = rowTile * 16;      // index into flattened B*N
  const int h0      = colTile * 16;      // output channel base
  const int m       = lane & 15;         // A-matrix row / D column-lane id
  const int hs      = lane >> 4;         // lane half select
  const int kbA     = hs * 8;            // A frag K offset per ISA layout
  const int kbB     = hs * 16;           // unused: B loads use kbA, same k order as A

  const float* xrow = x  + (size_t)(row0 + m) * CC_;
  const float* wrow = w1 + (size_t)(h0  + m) * CC_;   // n == lane&15 == m

  v8f acc = {};
#pragma unroll
  for (int kc = 0; kc < 3; ++kc) {                    // K = 96 = 3 * 32
    const int k0 = kc * 32;
    float fa[16], fb[16];
#pragma unroll
    for (int j = 0; j < 8; ++j) fa[j]     = xrow[k0 + kbA + j];
#pragma unroll
    for (int j = 0; j < 8; ++j) fa[8 + j] = xrow[k0 + kbA + 16 + j];
#pragma unroll
    for (int j = 0; j < 8; ++j) fb[j]     = wrow[k0 + kbA + j];
#pragma unroll
    for (int j = 0; j < 8; ++j) fb[8 + j] = wrow[k0 + kbA + 16 + j];
    v16h a = pack16(fa), b = pack16(fb);
    acc = __builtin_amdgcn_wmma_f32_16x16x32_f16(
        false, a, false, b, (short)0, acc, false, false);
  }

  // Per-lane D column is fixed: hcol = h0 + (lane&15)
  const int hcol = h0 + m;
  const float sc = g[hcol] * rsqrtf(var[hcol] + EPSV);
  const float sh = be[hcol] - mu[hcol] * sc;
  const int   b_ = row0 / NN_;
  const int   n0 = row0 % NN_;           // 4096 % 16 == 0: tile never straddles b
  float* dst = h_t + ((size_t)b_ * HH_ + hcol) * NN_ + n0;
#pragma unroll
  for (int r = 0; r < 8; ++r) {
    const int mm = r + hs * 8;
    const float v = fmaf(acc[r], sc, sh);
    dst[mm] = v > 0.f ? v : 0.f;
  }
}

// ---------------------------------------------------------------------------
// Kernel 2: per-(b,h) 4096-pt FFT (ortho) -> diagonal complex gating + ReLU ->
// inverse FFT (ortho), real part. In-place on h_t (each block owns one
// contiguous sequence). DIF forward leaves bit-reversed order; gating is
// elementwise with per-channel constants so it commutes with the permutation;
// DIT inverse consumes bit-reversed input -> natural output. Twiddle table
// built once per block in LDS to keep transcendentals off the critical path.
// ---------------------------------------------------------------------------
__global__ __launch_bounds__(256)
void fft_gate_ifft(float* __restrict__ data, const float* __restrict__ rmat,
                   const float* __restrict__ imat, const float* __restrict__ rb,
                   const float* __restrict__ ib) {
  __shared__ float re[NN_];
  __shared__ float im[NN_];
  __shared__ float twr[NN_ / 2];
  __shared__ float twi[NN_ / 2];

  const int tid = threadIdx.x;
  const int seq = blockIdx.x;
  const int hch = seq % HH_;
  float* base = data + (size_t)seq * NN_;

  for (int t = tid; t < NN_ / 2; t += 256) {
    float s, c;
    __sincosf(-6.283185307179586f * (float)t / (float)NN_, &s, &c);
    twr[t] = c; twi[t] = s;                      // W_4096^t
  }
  for (int t = tid; t < NN_; t += 256) { re[t] = base[t]; im[t] = 0.f; }
  __syncthreads();

  // Forward DIF radix-2: stage s = 12..1, twiddle W^(j << (12-s))
  for (int s = 12; s >= 1; --s) {
    const int half = 1 << (s - 1);
    const int tsh  = 12 - s;
    for (int idx = tid; idx < NN_ / 2; idx += 256) {
      const int j = idx & (half - 1);
      const int p = ((idx >> (s - 1)) << s) + j;
      const int q = p + half;
      const float ur = re[p], ui = im[p], vr = re[q], vi = im[q];
      re[p] = ur + vr; im[p] = ui + vi;
      const float tr = ur - vr, ti = ui - vi;
      const float cw = twr[j << tsh], sw = twi[j << tsh];
      re[q] = tr * cw - ti * sw;
      im[q] = tr * sw + ti * cw;
    }
    __syncthreads();
  }

  // Spectral gating (bit-reversed order is fine: elementwise, per-channel)
  const float rd  = rmat[hch * HH_ + hch];       // diagonal of r
  const float id  = imat[hch * HH_ + hch];       // diagonal of i
  const float rbv = rb[hch], ibv = ib[hch];
  const float fsc = 0.015625f;                   // 1/sqrt(4096) ortho scale
  for (int t = tid; t < NN_; t += 256) {
    const float xr = re[t] * fsc, xi = im[t] * fsc;
    const float gr = xr * rd - xi * id + rbv;
    const float gi = xi * rd + xr * id + ibv;
    re[t] = gr > 0.f ? gr : 0.f;
    im[t] = gi > 0.f ? gi : 0.f;
  }
  __syncthreads();

  // Inverse DIT radix-2: stage s = 1..12, twiddle conj(W)^(j << (12-s))
  for (int s = 1; s <= 12; ++s) {
    const int half = 1 << (s - 1);
    const int tsh  = 12 - s;
    for (int idx = tid; idx < NN_ / 2; idx += 256) {
      const int j = idx & (half - 1);
      const int p = ((idx >> (s - 1)) << s) + j;
      const int q = p + half;
      const float cw = twr[j << tsh], sw = -twi[j << tsh];
      const float vr = re[q] * cw - im[q] * sw;
      const float vi = re[q] * sw + im[q] * cw;
      const float ur = re[p], ui = im[p];
      re[p] = ur + vr; im[p] = ui + vi;
      re[q] = ur - vr; im[q] = ui - vi;
    }
    __syncthreads();
  }
  for (int t = tid; t < NN_; t += 256) base[t] = re[t] * fsc;  // real part, ortho
}

// ---------------------------------------------------------------------------
// Kernel 3: fc2 GEMM ([B*N,H] x [C,H]^T) + BN2 -> d_out [B,N,C].
// A is read from the transposed y_t[B,H,N]; per-k loads are two coalesced
// 64B half-wave segments. 6x v_wmma_f32_16x16x32_f16 per tile.
// ---------------------------------------------------------------------------
__global__ __launch_bounds__(256)
void gemm2_bn(const float* __restrict__ y_t, const float* __restrict__ w2,
              const float* __restrict__ g, const float* __restrict__ be,
              const float* __restrict__ mu, const float* __restrict__ var,
              float* __restrict__ out) {
  const int lane    = threadIdx.x & 31;
  const int wg      = blockIdx.x * 8 + (threadIdx.x >> 5);
  const int rowTile = wg / (CC_ / 16);
  const int colTile = wg % (CC_ / 16);
  const int row0    = rowTile * 16;
  const int c0      = colTile * 16;
  const int m       = lane & 15;
  const int hs      = lane >> 4;
  const int kbA     = hs * 8;
  const int kbB     = hs * 16;
  const int b_      = row0 / NN_;
  const int n0      = row0 % NN_;

  const float* acol = y_t + (size_t)b_ * HH_ * NN_ + n0 + m;  // + k*NN_
  const float* wrow = w2 + (size_t)(c0 + m) * HH_;            // n == lane&15

  v8f acc = {};
#pragma unroll
  for (int kc = 0; kc < 6; ++kc) {                 // K = 192 = 6 * 32
    const int k0 = kc * 32;
    float fa[16], fb[16];
#pragma unroll
    for (int j = 0; j < 8; ++j) fa[j]     = acol[(size_t)(k0 + kbA + j) * NN_];
#pragma unroll
    for (int j = 0; j < 8; ++j) fa[8 + j] = acol[(size_t)(k0 + kbA + 16 + j) * NN_];
#pragma unroll
    for (int j = 0; j < 8; ++j) fb[j]     = wrow[k0 + kbA + j];
#pragma unroll
    for (int j = 0; j < 8; ++j) fb[8 + j] = wrow[k0 + kbA + 16 + j];
    v16h a = pack16(fa), b = pack16(fb);
    acc = __builtin_amdgcn_wmma_f32_16x16x32_f16(
        false, a, false, b, (short)0, acc, false, false);
  }

  const int ccol = c0 + m;
  const float sc = g[ccol] * rsqrtf(var[ccol] + EPSV);
  const float sh = be[ccol] - mu[ccol] * sc;
  float* dst = out + (size_t)row0 * CC_ + ccol;
#pragma unroll
  for (int r = 0; r < 8; ++r) {
    const int mm = r + hs * 8;
    dst[(size_t)mm * CC_] = fmaf(acc[r], sc, sh);
  }
}

// ---------------------------------------------------------------------------
extern "C" void kernel_launch(void* const* d_in, const int* in_sizes, int n_in,
                              void* d_out, int out_size, void* d_ws, size_t ws_size,
                              hipStream_t stream) {
  const float* x         = (const float*)d_in[0];
  const float* w1        = (const float*)d_in[1];
  const float* bn1_gamma = (const float*)d_in[2];
  const float* bn1_beta  = (const float*)d_in[3];
  const float* bn1_mean  = (const float*)d_in[4];
  const float* bn1_var   = (const float*)d_in[5];
  const float* rmat      = (const float*)d_in[6];
  const float* imat      = (const float*)d_in[7];
  const float* rb        = (const float*)d_in[8];
  const float* ib        = (const float*)d_in[9];
  const float* w2        = (const float*)d_in[10];
  const float* bn2_gamma = (const float*)d_in[11];
  const float* bn2_beta  = (const float*)d_in[12];
  const float* bn2_mean  = (const float*)d_in[13];
  const float* bn2_var   = (const float*)d_in[14];
  float* out = (float*)d_out;

  // Single 100.7 MB intermediate, reused in place by the FFT kernel:
  // h_t[B,H,N] fp32 = 32*192*4096*4 bytes.
  float* h_t = (float*)d_ws;

  const int tilesM = (BB_ * NN_) / 16;             // 8192
  const int g1 = tilesM * (HH_ / 16) / 8;          // 12288 blocks (8 waves each)
  const int g2 = tilesM * (CC_ / 16) / 8;          // 6144 blocks

  gemm1_bn_relu<<<g1, 256, 0, stream>>>(x, w1, bn1_gamma, bn1_beta,
                                        bn1_mean, bn1_var, h_t);
  fft_gate_ifft<<<BB_ * HH_, 256, 0, stream>>>(h_t, rmat, imat, rb, ib);
  gemm2_bn<<<g2, 256, 0, stream>>>(h_t, w2, bn2_gamma, bn2_beta,
                                   bn2_mean, bn2_var, out);
}
